// SelfAttention_7559142441747
// MI455X (gfx1250) — matrix operand order repaired and checked by reading, not compiled, not run on hardware
//
#include <hip/hip_runtime.h>
#include <hip/hip_bf16.h>
#include <math.h>

// ---------------------------------------------------------------------------
// Self-attention over grouped point clouds, CDNA5 (gfx1250, wave32, WMMA).
// B=4, M=8192, K=32, CH=64. One wave per (b,m) group; 4 waves (4 groups) per
// 128-thread block. f16 WMMA 16x16x32 for both GEMMs, f32 accumulate.
// ---------------------------------------------------------------------------

typedef __attribute__((ext_vector_type(16))) _Float16 v16h;
typedef __attribute__((ext_vector_type(8)))  float    v8f;

#define B_TOT 4
#define M_TOT 8192
#define KPTS  32
#define CHN   64

__device__ __forceinline__ void lds_fence() {
    asm volatile("s_wait_dscnt 0" ::: "memory");
}

// A-fragment (16x32 f16, D = A x B): lane holds row M = base_m + (lane&15).
// K pairs contiguous: VGPR i -> K = 2i (+16 for lanes 16..31).
// src is row-major [row][k] with row stride `stride` halves; kbase selects the
// 32-wide K chunk. All k indices are even -> single ds_load_b32 per VGPR.
__device__ __forceinline__ v16h load_frag_A(const _Float16* src, int base_m,
                                            int stride, int kbase, int lane) {
    union { v16h h; unsigned u[8]; } f;
    const int row  = base_m + (lane & 15);
    const int koff = (lane >> 4) ? 16 : 0;
#pragma unroll
    for (int i = 0; i < 8; ++i) {
        const int k = 2 * i + koff + kbase;
        f.u[i] = *(const unsigned*)(src + row * stride + k);
    }
    return f.h;
}

// B-fragment (32x16 f16): lane holds column N = base_n + (lane&15).
// VGPR i -> K = 2i (+16 for lanes 16..31). src is laid out [col][k] with
// stride `stride` halves (i.e. the K pairs are contiguous per column).
__device__ __forceinline__ v16h load_frag_B(const _Float16* src, int base_n,
                                            int stride, int kbase, int lane) {
    union { v16h h; unsigned u[8]; } f;
    const int col  = base_n + (lane & 15);
    const int koff = (lane >> 4) ? 16 : 0;
#pragma unroll
    for (int i = 0; i < 8; ++i) {
        const int k = 2 * i + koff + kbase;
        f.u[i] = *(const unsigned*)(src + col * stride + k);
    }
    return f.h;
}

__device__ __forceinline__ v8f wmma_f16(v16h a, v16h b, v8f c) {
    return __builtin_amdgcn_wmma_f32_16x16x32_f16(false, a, false, b,
                                                  (short)0, c, false, false);
}

__global__ __launch_bounds__(128)
void attn_pe_kernel(const float* __restrict__ xyz,
                    const float* __restrict__ Wq, const float* __restrict__ bq,
                    const float* __restrict__ Wk, const float* __restrict__ bk,
                    const float* __restrict__ Wv, const float* __restrict__ bv,
                    float* __restrict__ out) {
    // ---- static LDS ------------------------------------------------------
    __shared__ float sWq[192], sBq[64], sWk[192], sBk[64], sWv[192], sBv[64];
    __shared__ float sXyz[3][KPTS][4];                 // xyz tile for 4 groups
    __shared__ __align__(16) unsigned char sRegion1[4][4096]; // qh -> vt
    __shared__ __align__(16) unsigned char sRegion2[4][4096]; // kh -> attn
    __shared__ float sScore[4][KPTS * KPTS];           // f32 scores

    const int t    = threadIdx.x;
    const int lane = t & 31;
    const int w    = t >> 5;                  // wave id in block = group slot
    const int b    = blockIdx.x >> 11;        // 2048 blocks per batch element
    const int m0   = (blockIdx.x & 2047) << 2;
    const int m    = m0 + w;

    // ---- stage weights + biases (uniform, L2-hot) ------------------------
    for (int i = t; i < 192; i += 128) {
        sWq[i] = Wq[i]; sWk[i] = Wk[i]; sWv[i] = Wv[i];
    }
    if (t < 64) { sBq[t] = bq[t]; sBk[t] = bk[t]; sBv[t] = bv[t]; }

    // ---- stage xyz tile: [c][k][dm], dm = m-offset within block ----------
    for (int i = t; i < 3 * KPTS * 4; i += 128) {
        const int dm = i & 3;
        const int rk = i >> 2;
        const int k  = rk & 31;
        const int c  = rk >> 5;
        sXyz[c][k][dm] =
            xyz[((size_t)(b * 3 + c) * KPTS + k) * M_TOT + m0 + dm];
    }
    __syncthreads();

    _Float16* qh = (_Float16*)&sRegion1[w][0];   // [row][ch], 32x64
    _Float16* kh = (_Float16*)&sRegion2[w][0];   // [row][ch], 32x64
    float*    sc = &sScore[w][0];                // [row][j],  32x32

    const float x = sXyz[0][lane][w];
    const float y = sXyz[1][lane][w];
    const float z = sXyz[2][lane][w];

    // ---- q / k projection: lane = k-row, 3 FMAs per channel --------------
    unsigned* qh32 = (unsigned*)qh;
    unsigned* kh32 = (unsigned*)kh;
#pragma unroll
    for (int ch = 0; ch < CHN; ch += 2) {
        float q0 = fmaf(sWq[3*ch+2], z, fmaf(sWq[3*ch+1], y, fmaf(sWq[3*ch+0], x, sBq[ch])));
        float q1 = fmaf(sWq[3*ch+5], z, fmaf(sWq[3*ch+4], y, fmaf(sWq[3*ch+3], x, sBq[ch+1])));
        float k0 = fmaf(sWk[3*ch+2], z, fmaf(sWk[3*ch+1], y, fmaf(sWk[3*ch+0], x, sBk[ch])));
        float k1 = fmaf(sWk[3*ch+5], z, fmaf(sWk[3*ch+4], y, fmaf(sWk[3*ch+3], x, sBk[ch+1])));
        union { _Float16 h[2]; unsigned u; } pq, pk;
        pq.h[0] = (_Float16)q0; pq.h[1] = (_Float16)q1;
        pk.h[0] = (_Float16)k0; pk.h[1] = (_Float16)k1;
        qh32[lane * 32 + (ch >> 1)] = pq.u;
        kh32[lane * 32 + (ch >> 1)] = pk.u;
    }
    lds_fence();

    // ---- score = q k^T / sqrt(CH): 2x2 tiles, K=64 in two WMMA steps -----
#pragma unroll
    for (int r = 0; r < 2; ++r) {
        const v16h a0 = load_frag_A(qh, r * 16, CHN, 0,  lane);
        const v16h a1 = load_frag_A(qh, r * 16, CHN, 32, lane);
#pragma unroll
        for (int c = 0; c < 2; ++c) {
            const v16h b0 = load_frag_B(kh, c * 16, CHN, 0,  lane);
            const v16h b1 = load_frag_B(kh, c * 16, CHN, 32, lane);
            v8f acc = {};
            acc = wmma_f16(a0, b0, acc);
            acc = wmma_f16(a1, b1, acc);
            const int colN  = c * 16 + (lane & 15);
            const int rbase = r * 16 + ((lane >> 4) ? 8 : 0);
#pragma unroll
            for (int i = 0; i < 8; ++i)
                sc[(rbase + i) * KPTS + colN] = acc[i] * 0.125f; // 1/sqrt(64)
        }
    }
    lds_fence();

    // ---- v projection, stored transposed vt[ch][j] over dead q buffer ----
    _Float16* vt = (_Float16*)&sRegion1[w][0];   // [ch][j], 64x32
#pragma unroll
    for (int ch = 0; ch < CHN; ++ch) {
        float vv = fmaf(sWv[3*ch+2], z, fmaf(sWv[3*ch+1], y, fmaf(sWv[3*ch+0], x, sBv[ch])));
        vt[ch * KPTS + lane] = (_Float16)vv;
    }

    // ---- row softmax: lane = row ----------------------------------------
    float sv[KPTS];
#pragma unroll
    for (int j = 0; j < KPTS; ++j) sv[j] = sc[lane * KPTS + j];
    float mx = sv[0];
#pragma unroll
    for (int j = 1; j < KPTS; ++j) mx = fmaxf(mx, sv[j]);
    float sum = 0.0f;
#pragma unroll
    for (int j = 0; j < KPTS; ++j) { sv[j] = __expf(sv[j] - mx); sum += sv[j]; }
    const float inv = 1.0f / sum;

    _Float16* attn = (_Float16*)&sRegion2[w][0]; // [row][j], 32x32 (over kh)
    unsigned* at32 = (unsigned*)attn;
#pragma unroll
    for (int j = 0; j < KPTS; j += 2) {
        union { _Float16 h[2]; unsigned u; } pa;
        pa.h[0] = (_Float16)(sv[j] * inv);
        pa.h[1] = (_Float16)(sv[j + 1] * inv);
        at32[lane * 16 + (j >> 1)] = pa.u;
    }
    lds_fence();

    // ---- pe = relu(attn @ v): 2x4 tiles, single K=32 WMMA each -----------
    const size_t outBase = (size_t)(b * M_TOT + m) * (KPTS * CHN);
#pragma unroll
    for (int r = 0; r < 2; ++r) {
        const v16h af = load_frag_A(attn, r * 16, KPTS, 0, lane);
#pragma unroll
        for (int c = 0; c < 4; ++c) {
            const v16h bf = load_frag_B(vt, c * 16, KPTS, 0, lane);
            v8f acc = {};
            acc = wmma_f16(af, bf, acc);
            const int colN  = c * 16 + (lane & 15);
            const int rbase = r * 16 + ((lane >> 4) ? 8 : 0);
#pragma unroll
            for (int i = 0; i < 8; ++i)
                out[outBase + (size_t)(rbase + i) * CHN + colN] =
                    fmaxf(acc[i], 0.0f);
        }
    }
}

extern "C" void kernel_launch(void* const* d_in, const int* in_sizes, int n_in,
                              void* d_out, int out_size, void* d_ws, size_t ws_size,
                              hipStream_t stream) {
    const float* xyz = (const float*)d_in[0];
    const float* Wq  = (const float*)d_in[1];
    const float* bq  = (const float*)d_in[2];
    const float* Wk  = (const float*)d_in[3];
    const float* bk  = (const float*)d_in[4];
    const float* Wv  = (const float*)d_in[5];
    const float* bv  = (const float*)d_in[6];
    float* out = (float*)d_out;

    // B*M / 4 groups-per-block = 4*8192/4 = 8192 blocks of 128 threads.
    attn_pe_kernel<<<dim3(8192), dim3(128), 0, stream>>>(
        xyz, Wq, bq, Wk, bk, Wv, bv, out);
}
